// MatchLoss_83863531421817
// MI455X (gfx1250) — matrix operand order repaired and checked by reading, not compiled, not run on hardware
//
#include <hip/hip_runtime.h>

typedef __attribute__((ext_vector_type(16))) _Float16 v16h;
typedef __attribute__((ext_vector_type(8)))  _Float16 v8h;
typedef __attribute__((ext_vector_type(8)))  float    v8f;

#define BATCH 2
#define CHN   128
#define NPOS  4096          // 64*64 spatial positions
#define L1EPS 1e-12f

#define BM    256           // output block rows
#define BN    256           // output block cols
#define LDSH  136           // LDS row stride in halves (272 B = 17*16B, conflict pad)
#define GEMM_THREADS 512    // 16 waves
#define GEMM_BLOCKS  (BATCH * (NPOS / BM) * (NPOS / BN))   // 512
#define LDS_BYTES    ((unsigned)(BN * LDSH * sizeof(_Float16)))  // 69632 (B block only)

// ---------------------------------------------------------------------------
// Pass 1: L1-normalize along channels, form D = norm(c) - norm(b), pack A'
// and D' as f16 position-major (N x C). One thread per (batch, position).
// ---------------------------------------------------------------------------
__global__ __launch_bounds__(256) void norm_pack_kernel(
    const float* __restrict__ a, const float* __restrict__ b,
    const float* __restrict__ c,
    _Float16* __restrict__ An, _Float16* __restrict__ Dn)
{
  int pos = blockIdx.x * blockDim.x + threadIdx.x;   // 0 .. BATCH*NPOS-1
  int bt  = pos / NPOS;
  int n   = pos - bt * NPOS;
  size_t base = (size_t)bt * CHN * NPOS + n;

  float na = 0.f, nb = 0.f, nc = 0.f;
  for (int ch = 0; ch < CHN; ++ch) {
    size_t idx = base + (size_t)ch * NPOS;
    na += fabsf(a[idx]);
    nb += fabsf(b[idx]);
    nc += fabsf(c[idx]);
  }
  float ia = 1.0f / fmaxf(na, L1EPS);
  float ib = 1.0f / fmaxf(nb, L1EPS);
  float ic = 1.0f / fmaxf(nc, L1EPS);

  size_t obase = (size_t)pos * CHN;
  for (int ch = 0; ch < CHN; ++ch) {
    size_t idx = base + (size_t)ch * NPOS;
    An[obase + ch] = (_Float16)(a[idx] * ia);
    Dn[obase + ch] = (_Float16)(c[idx] * ic - b[idx] * ib);
  }
}

// ---------------------------------------------------------------------------
// Async-copy one contiguous 256-row x 256-byte operand block (64 KB) into LDS
// (272-byte padded rows) via CDNA5 global_load_async_to_lds_b128, GVS mode:
// per-lane LDS byte offset in VDST, per-lane global byte offset in VADDR,
// uniform 64-bit base in an SGPR pair. Tracked by ASYNCcnt.
// ---------------------------------------------------------------------------
__device__ __forceinline__ void async_copy_block(unsigned lds_base,
                                                 const _Float16* gbase,
                                                 int tid)
{
  const unsigned chunk = (unsigned)(tid & 15) * 16u;   // 16-byte chunk in row
  const unsigned row0  = (unsigned)(tid >> 4);         // 0..31
  const unsigned long long saddr = (unsigned long long)(uintptr_t)gbase;
#pragma unroll
  for (int p = 0; p < 8; ++p) {
    unsigned row    = row0 + (unsigned)p * 32u;        // 0..255
    unsigned ldsoff = lds_base + row * 272u + chunk;
    unsigned goff   = row * 256u + chunk;
    asm volatile("global_load_async_to_lds_b128 %0, %1, %2 offset:0"
                 :: "v"(ldsoff), "v"(goff), "s"(saddr) : "memory");
  }
}

// ---------------------------------------------------------------------------
// Pass 2: S = A' * D'^T per batch (4096x4096, K=128) with
// v_wmma_f32_16x16x32_f16, reducing sum(|S|) in registers.
//
// Block = 16 waves, 256x256 output. D' (B-operand) block async-copied to LDS
// once (it is shared by all waves); A' rows stream from global into registers
// (contiguous per lane, reused 8x). Wave w = (rt2 = w&7, cg = w>>3) computes
// 2 row-tiles x 8 col-tiles: each B fragment feeds two WMMAs, and the next
// tile's B fragments are prefetched into separate registers so ds_load
// latency overlaps the WMMAs.
//
// Fragment layouts used here (f16, contiguous K order in both A and B):
//   A (16x32): lanes 0-15 = row M, 16 consecutive K from 0;
//              lanes 16-31 = row M, 16 consecutive K from 16.
//   B (32x16): lanes 0-15 = col N, 16 consecutive K from 0;
//              lanes 16-31 = col N, 16 consecutive K from 16.
// ---------------------------------------------------------------------------
__global__ __launch_bounds__(GEMM_THREADS) void gemm_absum_kernel(
    const _Float16* __restrict__ An, const _Float16* __restrict__ Dn,
    float* __restrict__ partials)
{
  extern __shared__ _Float16 smem[];
  _Float16* Bs = smem;               // [BN][LDSH]

  const int tid  = threadIdx.x;
  const int wid  = tid >> 5;         // wave 0..15
  const int lane = tid & 31;

  const int bt  = blockIdx.x >> 8;          // batch
  const int rem = blockIdx.x & 255;
  const int bm  = rem >> 4;                 // row block 0..15
  const int bn  = rem & 15;                 // col block 0..15

  const _Float16* Ab = An + (size_t)bt * NPOS * CHN + (size_t)bm * BM * CHN;
  const _Float16* Db = Dn + (size_t)bt * NPOS * CHN + (size_t)bn * BN * CHN;

  // ---- async copy the B (D') block into LDS ----------------------------
  const unsigned bs_base = (unsigned)(uintptr_t)(void*)Bs;
  async_copy_block(bs_base, Db, tid);

  // ---- A fragments: 2 row-tiles, all 4 K-steps, straight from global ---
  const int rt2   = wid & 7;            // row pair 0..7  -> rows rt2*32..+31
  const int cg    = wid >> 3;           // col group 0..1 -> cols cg*128..+127
  const int akoff = (lane >> 4) * 16;   // halves
  const int bkoff = (lane >> 4) * 16;   // halves

  v16h afrag[2][4];
#pragma unroll
  for (int rti = 0; rti < 2; ++rti) {
    const int row = rt2 * 32 + rti * 16 + (lane & 15);
    const _Float16* arow = Ab + (size_t)row * CHN + akoff;
#pragma unroll
    for (int k = 0; k < 4; ++k) {
      union { v16h v; v8h h[2]; } u;
      u.h[0] = *(const v8h*)(arow + k * 32);
      u.h[1] = *(const v8h*)(arow + k * 32 + 8);
      afrag[rti][k] = u.v;
    }
  }

  asm volatile("s_wait_asynccnt 0x0" ::: "memory");
  __syncthreads();

  // ---- pipelined B fetch + WMMA ---------------------------------------
  const _Float16* bcol0 = Bs + bkoff;
  auto loadB = [&](int ct, v16h* dst) {
    const _Float16* bp = bcol0 + (size_t)(ct * 16 + (lane & 15)) * LDSH;
#pragma unroll
    for (int k = 0; k < 4; ++k) {
      union { v16h v; v8h h[2]; } u;
      u.h[0] = *(const v8h*)(bp + k * 32);
      u.h[1] = *(const v8h*)(bp + k * 32 + 8);
      dst[k] = u.v;
    }
  };

  v16h bfrag[4], bnext[4];
  loadB(cg * 8, bfrag);

  v8f asum = {};
#pragma unroll
  for (int c8 = 0; c8 < 8; ++c8) {
    if (c8 < 7) loadB(cg * 8 + c8 + 1, bnext);   // prefetch next tile's B

#pragma unroll
    for (int rti = 0; rti < 2; ++rti) {
      v8f acc = {};
#pragma unroll
      for (int k = 0; k < 4; ++k)
        acc = __builtin_amdgcn_wmma_f32_16x16x32_f16(false, afrag[rti][k],
                                                     false, bfrag[k],
                                                     (short)0, acc,
                                                     false, false);
#pragma unroll
      for (int i = 0; i < 8; ++i)       // 8 independent adds, no serial chain
        asum[i] += fabsf(acc[i]);
    }

#pragma unroll
    for (int k = 0; k < 4; ++k) bfrag[k] = bnext[k];
  }

  float local = ((asum[0] + asum[4]) + (asum[1] + asum[5])) +
                ((asum[2] + asum[6]) + (asum[3] + asum[7]));

  // ---- wave32 + block reduction ---------------------------------------
#pragma unroll
  for (int off = 16; off > 0; off >>= 1)
    local += __shfl_xor(local, off, 32);

  __shared__ float sred[16];
  if (lane == 0) sred[wid] = local;
  __syncthreads();
  if (tid == 0) {
    float s = 0.f;
#pragma unroll
    for (int i = 0; i < 16; ++i) s += sred[i];
    partials[blockIdx.x] = s;
  }
}

// ---------------------------------------------------------------------------
// Pass 3: reduce per-block partials, apply mean scale. Single block.
// ---------------------------------------------------------------------------
__global__ __launch_bounds__(256) void final_reduce_kernel(
    const float* __restrict__ partials, int n, float* __restrict__ out)
{
  __shared__ float s[256];
  float local = 0.f;
  for (int i = threadIdx.x; i < n; i += 256) local += partials[i];
  s[threadIdx.x] = local;
  __syncthreads();
  for (int off = 128; off > 0; off >>= 1) {
    if (threadIdx.x < off) s[threadIdx.x] += s[threadIdx.x + off];
    __syncthreads();
  }
  if (threadIdx.x == 0)
    out[0] = s[0] * (1.0f / ((float)BATCH * NPOS * NPOS));  // mean over 2*4096^2
}

// ---------------------------------------------------------------------------
extern "C" void kernel_launch(void* const* d_in, const int* in_sizes, int n_in,
                              void* d_out, int out_size, void* d_ws, size_t ws_size,
                              hipStream_t stream)
{
  const float* a = (const float*)d_in[0];
  const float* b = (const float*)d_in[1];
  const float* c = (const float*)d_in[2];

  _Float16* An = (_Float16*)d_ws;                                // 2 MB
  _Float16* Dn = An + (size_t)BATCH * NPOS * CHN;                // 2 MB
  float* partials = (float*)(Dn + (size_t)BATCH * NPOS * CHN);   // 2 KB

  norm_pack_kernel<<<(BATCH * NPOS) / 256, 256, 0, stream>>>(a, b, c, An, Dn);

  gemm_absum_kernel<<<GEMM_BLOCKS, GEMM_THREADS, LDS_BYTES, stream>>>(An, Dn,
                                                                      partials);

  final_reduce_kernel<<<1, 256, 0, stream>>>(partials, GEMM_BLOCKS,
                                             (float*)d_out);
}
